// DisRNNCellNet_59742995088100
// MI455X (gfx1250) — matrix operand order repaired and checked by reading, not compiled, not run on hardware
//
#include <hip/hip_runtime.h>

typedef _Float16 v16h __attribute__((ext_vector_type(16)));
typedef _Float16 v8h  __attribute__((ext_vector_type(8)));
typedef float    v8f  __attribute__((ext_vector_type(8)));
typedef float    v4f  __attribute__((ext_vector_type(4)));

#define TSTEP 32
#define TBSIZE 4096
#define TNNOD 8
#define TDIM 64
#define THID 64
#define NGATE 256                 // 4*HID
#define TBN (TBSIZE * TNNOD)      // 32768
#define WAVES 8
#define MTILES 2                  // two 16-row M-tiles per wave
#define ROWS_PER_WAVE (16 * MTILES)             // 32
#define ROWS_PER_BLOCK (WAVES * ROWS_PER_WAVE)  // 256

__device__ __forceinline__ float rcp_fast(float x) { return __builtin_amdgcn_rcpf(x); }
__device__ __forceinline__ float sigf(float x) { return rcp_fast(1.0f + __expf(-x)); }
__device__ __forceinline__ float tanh_fast(float x) {
    return __builtin_fmaf(2.0f, rcp_fast(1.0f + __expf(-2.0f * x)), -1.0f);
}

// pack one x row (f32) into the two K-half A-fragments for this lane
__device__ __forceinline__ void pack_a_frags(const float* __restrict__ row, int hi,
                                             v16h& a0, v16h& a1) {
    const v4f* p = (const v4f*)row;
    v4f c0 = p[hi * 4 + 0],      c1 = p[hi * 4 + 1];       // K = hi*16 ..
    v4f c2 = p[hi * 4 + 2],      c3 = p[hi * 4 + 3];       // K = hi*16+8 ..
    v4f d0 = p[8 + hi * 4 + 0],  d1 = p[8 + hi * 4 + 1];   // K = 32+hi*16 ..
    v4f d2 = p[8 + hi * 4 + 2],  d3 = p[8 + hi * 4 + 3];   // K = 40+hi*16 ..
#pragma unroll
    for (int e = 0; e < 4; ++e) {
        a0[e]     = (_Float16)c0[e]; a0[4 + e]  = (_Float16)c1[e];
        a0[8 + e] = (_Float16)c2[e]; a0[12 + e] = (_Float16)c3[e];
        a1[e]     = (_Float16)d0[e]; a1[4 + e]  = (_Float16)d1[e];
        a1[8 + e] = (_Float16)d2[e]; a1[12 + e] = (_Float16)d3[e];
    }
}

__global__ void __launch_bounds__(256, 1)
tdlstm_kernel(const float* __restrict__ input,     // [S, BN, 64]
              const float* __restrict__ time_dis,  // [B, S]
              const float* __restrict__ w_ih,      // [256, 64]
              const float* __restrict__ w_hh,      // [256, 64]
              const float* __restrict__ b_ih,      // [256]
              const float* __restrict__ b_hh,      // [256]
              const float* __restrict__ fc2_w,     // [64]
              const float* __restrict__ fc2_b,     // [1]
              const float* __restrict__ baseline,  // [1]
              float* __restrict__ out)             // [B]
{
    // weights row-major f16: B-fragment for column n, K-half kh is the
    // contiguous 16 halves at [n*64 + kh*32 + hi*16]  (32B aligned)
    __shared__ _Float16 wih_lds[NGATE * TDIM];                    // 32 KB
    __shared__ _Float16 whh_lds[NGATE * TDIM];                    // 32 KB
    __shared__ _Float16 h_lds[WAVES][ROWS_PER_WAVE][THID];        // 32 KB
    __shared__ float    om_lds[WAVES][ROWS_PER_WAVE];             // 1 KB

    const int tid  = threadIdx.x;
    const int wave = tid >> 5;
    const int lane = tid & 31;
    const int lrow = lane & 15;
    const int hi   = lane >> 4;

    // stage weights f32 -> f16 (coalesced reads)
    for (int idx = tid; idx < NGATE * TDIM; idx += 256) {
        wih_lds[idx] = (_Float16)w_ih[idx];
        whh_lds[idx] = (_Float16)w_hh[idx];
    }
    // zero this wave's h tile (h0 = 0)
    for (int j = lane; j < ROWS_PER_WAVE * THID; j += 32)
        (&h_lds[wave][0][0])[j] = (_Float16)0.0f;
    __syncthreads();

    const int m0 = blockIdx.x * ROWS_PER_BLOCK + wave * ROWS_PER_WAVE;

    // combined gate bias per N-tile (lane's column = nt*16 + lrow)
    float bias[16];
#pragma unroll
    for (int nt = 0; nt < 16; ++nt) {
        int n = nt * 16 + lrow;
        bias[nt] = b_ih[n] + b_hh[n];
    }

    // cell state, D-layout: cst[t][ct][e] = c[row t*16 + e+8*hi][col ct*16+lrow]
    v8f cst[MTILES][4];
#pragma unroll
    for (int t = 0; t < MTILES; ++t)
#pragma unroll
        for (int ct = 0; ct < 4; ++ct)
#pragma unroll
            for (int e = 0; e < 8; ++e) cst[t][ct][e] = 0.0f;

    for (int s = 0; s < TSTEP; ++s) {
        // ---- x / h A-fragments for both M-tiles ----
        v16h xa[MTILES][2], ha[MTILES][2];
#pragma unroll
        for (int t = 0; t < MTILES; ++t) {
            const float* xrow = input + ((size_t)s * TBN + m0 + t * 16 + lrow) * TDIM;
            pack_a_frags(xrow, hi, xa[t][0], xa[t][1]);
            if (s + 1 < TSTEP)
                __builtin_prefetch(xrow + (size_t)TBN * TDIM, 0, 0);  // next step's row
            // h A-fragments from wave-private LDS
            const v8h* hp = (const v8h*)(&h_lds[wave][t * 16 + lrow][0]);
            union { v16h v; v8h h2[2]; } u0, u1;
            u0.h2[0] = hp[2 * hi];     u0.h2[1] = hp[2 * hi + 1];
            u1.h2[0] = hp[4 + 2 * hi]; u1.h2[1] = hp[5 + 2 * hi];
            ha[t][0] = u0.v; ha[t][1] = u1.v;
        }

        // per-lane decay (constant across a lane's 8 D rows of each M-tile)
        float decay[MTILES];
#pragma unroll
        for (int t = 0; t < MTILES; ++t) {
            int cb = (m0 >> 3) + t * 2 + hi;
            float td = time_dis[cb * TSTEP + (s == 0 ? 0 : s - 1)];
            decay[t] = rcp_fast(__logf(2.718281828459045f + td));
        }

        // ---- 16 tile-jobs: j -> ct = j>>2, gate = j&3, ntile = gate*4+ct ----
        // Double-buffered B-fragment pipeline; each fragment feeds MTILES WMMAs.
        // Accumulators start from WMMA's inline-0 C operand; bias is added in
        // the elementwise stage (cheaper than splatting bias into 8 VGPRs).
        v16h fra[4], frb[4];
        auto loadfr = [&](int j, v16h* dst) {
            int nt = (j & 3) * 4 + (j >> 2);
            int n  = nt * 16 + lrow;
            dst[0] = *(const v16h*)&wih_lds[n * TDIM + 0  + hi * 16];
            dst[1] = *(const v16h*)&wih_lds[n * TDIM + 32 + hi * 16];
            dst[2] = *(const v16h*)&whh_lds[n * TDIM + 0  + hi * 16];
            dst[3] = *(const v16h*)&whh_lds[n * TDIM + 32 + hi * 16];
        };
        loadfr(0, fra);
        v8f ga[MTILES][4];
#pragma unroll
        for (int j = 0; j < 16; ++j) {
            const int ct = j >> 2;
            const int g  = j & 3;
            v16h* cur = (j & 1) ? frb : fra;
            v16h* nxt = (j & 1) ? fra : frb;
            if (j + 1 < 16) loadfr(j + 1, nxt);

#pragma unroll
            for (int t = 0; t < MTILES; ++t) {
                v8f a = {};   // lowers to inline 0 C operand on the first WMMA
                a = __builtin_amdgcn_wmma_f32_16x16x32_f16(false, xa[t][0], false, cur[0], (short)0, a, false, false);
                a = __builtin_amdgcn_wmma_f32_16x16x32_f16(false, xa[t][1], false, cur[1], (short)0, a, false, false);
                a = __builtin_amdgcn_wmma_f32_16x16x32_f16(false, ha[t][0], false, cur[2], (short)0, a, false, false);
                a = __builtin_amdgcn_wmma_f32_16x16x32_f16(false, ha[t][1], false, cur[3], (short)0, a, false, false);
                ga[t][g] = a;
            }

            if (g == 3) {   // all four gates of column-tile ct ready -> update c, h
#pragma unroll
                for (int t = 0; t < MTILES; ++t) {
#pragma unroll
                    for (int e = 0; e < 8; ++e) {
                        float ig = sigf(ga[t][0][e] + bias[ct]);
                        float fg = sigf(ga[t][1][e] + bias[4 + ct]);
                        float gg = tanh_fast(ga[t][2][e] + bias[8 + ct]);
                        float og = sigf(ga[t][3][e] + bias[12 + ct]);
                        float cn = __builtin_fmaf(fg * decay[t], cst[t][ct][e], ig * gg);
                        cst[t][ct][e] = cn;
                        h_lds[wave][t * 16 + e + 8 * hi][ct * 16 + lrow] =
                            (_Float16)(og * tanh_fast(cn));
                    }
                }
            }
        }
        // compiler fence only: same-wave LDS ops execute in order in HW, so the
        // next step's h loads see this step's h stores without a counter drain.
        asm volatile("" ::: "memory");
    }

    // ---- readout: out2 = sigmoid(h . fc2 + b); noisy-OR over 8 nodules ----
    float base_prob = sigf(baseline[0]);
    {
        // each lane reduces one of the wave's 32 rows
        const _Float16* hrow = &h_lds[wave][lane][0];
        float dot = fc2_b[0];
#pragma unroll
        for (int j = 0; j < THID; ++j) dot += (float)hrow[j] * fc2_w[j];
        om_lds[wave][lane] = 1.0f - sigf(dot);
    }
    asm volatile("s_wait_dscnt 0x0" ::: "memory");
    if ((lane & 7) == 0) {      // 4 cases per wave
        float p = 1.0f;
#pragma unroll
        for (int k = 0; k < 8; ++k) p *= om_lds[wave][(lane & 24) + k];
        out[(m0 >> 3) + (lane >> 3)] = 1.0f - p * (1.0f - base_prob);
    }
}

extern "C" void kernel_launch(void* const* d_in, const int* in_sizes, int n_in,
                              void* d_out, int out_size, void* d_ws, size_t ws_size,
                              hipStream_t stream) {
    const float* input    = (const float*)d_in[0];
    const float* time_dis = (const float*)d_in[1];
    const float* w_ih     = (const float*)d_in[2];
    const float* w_hh     = (const float*)d_in[3];
    const float* b_ih     = (const float*)d_in[4];
    const float* b_hh     = (const float*)d_in[5];
    const float* fc2_w    = (const float*)d_in[6];
    const float* fc2_b    = (const float*)d_in[7];
    const float* baseline = (const float*)d_in[8];

    dim3 grid(TBN / ROWS_PER_BLOCK);   // 128 blocks of 256 rows
    dim3 block(256);                   // 8 waves (wave32), 32 rows each
    hipLaunchKernelGGL(tdlstm_kernel, grid, block, 0, stream,
                       input, time_dis, w_ih, w_hh, b_ih, b_hh,
                       fc2_w, fc2_b, baseline, (float*)d_out);
    (void)d_ws; (void)ws_size; (void)in_sizes; (void)n_in; (void)out_size;
}
